// SuperGlue_52819507806319
// MI455X (gfx1250) — matrix operand order repaired and checked by reading, not compiled, not run on hardware
//
#include <hip/hip_runtime.h>
#include <cstdint>
#include <cstddef>

// ---------------------------------------------------------------------------
// SuperGlue-style pipeline for MI455X (gfx1250, wave32, WMMA).
// All GEMMs go through one tiled v_wmma_f32_16x16x32_f16 kernel (f32 accum).
// A-tile staging uses gfx1250 async global->LDS copies (ASYNCcnt).
// Workspace requirement: ~80 MB.
// ---------------------------------------------------------------------------

typedef __attribute__((ext_vector_type(16))) _Float16 v16h;
typedef __attribute__((ext_vector_type(8)))  _Float16 h8;
typedef __attribute__((ext_vector_type(8)))  float    v8f;

namespace cfg {
constexpr int N   = 2048;       // keypoints per image (N == M)
constexpr int D   = 256;
constexpr int H   = 4;
constexpr int HD  = 64;
constexpr int L   = 10;
constexpr int SN  = 2049;       // N + 1 (dustbin)
constexpr float NORM  = -8.317766166719343f;  // -log(4096)
constexpr float LOG_N =  7.624618986159398f;  // log(2048)
}

// ============================= WMMA GEMM ===================================
// C[M,N] = alpha * A[M,K] @ B[K,N] (+ bias per row).  Row-major, f16 in,
// f32 out.  M % 128 == 0, N % 64 == 0, K % 32 == 0, lda/ldb % 8 == 0
// (guaranteed by caller).
#define TM 128
#define TN 64
#define TK 32
#define LDS_AS 40   // padded row stride (halves) for A tile
#define LDS_BS 40   // padded row stride (halves) for transposed B tile

__global__ __launch_bounds__(256) void gemm_wmma_f16(
    const _Float16* __restrict__ A, int lda,
    const _Float16* __restrict__ B, int ldb,
    float* __restrict__ C, int ldc,
    int M, int N, int K, float alpha,
    const float* __restrict__ bias)
{
  __shared__ __align__(16) _Float16 As[TM * LDS_AS];       // As[row][k]
  __shared__ __align__(16) _Float16 Bs[TN * LDS_BS];       // Bs[col][k] (transposed)

  const int tid  = threadIdx.x;
  const int lane = tid & 31;
  const int wave = tid >> 5;           // 8 waves
  const int wm   = wave & 3;           // wave row group (4)
  const int wn   = wave >> 2;          // wave col group (2)
  const int brow = blockIdx.y * TM;
  const int bcol = blockIdx.x * TN;

  const int laneHalf = lane >> 4;      // 0/1
  const int laneIdx  = lane & 15;
  const int kbase    = laneHalf * 8;   // A-operand K base per ISA layout

  union Frag { v16h v; h8 q[2]; };

  v8f acc[2][2] = {};

  for (int k0 = 0; k0 < K; k0 += TK) {
    // ---- stage A tile (128 x 32) f16: async global->LDS (32B/thread) ----
    {
      const int r   = tid >> 1;                 // 0..127
      const int seg = (tid & 1) * 16;           // 0 / 16
      const _Float16* src = A + (size_t)(brow + r) * lda + (k0 + seg);
      _Float16* dst = As + r * LDS_AS + seg;
#ifdef __gfx1250__
      // Low 32 bits of a generic LDS pointer are the wave-relative LDS byte
      // offset (ISA 10.2 aperture rules); INST_OFFSET applies to both sides.
      uint32_t dst_off = (uint32_t)(uintptr_t)dst;
      asm volatile("global_load_async_to_lds_b128 %0, %1, off"
                   :: "v"(dst_off), "v"(src) : "memory");
      asm volatile("global_load_async_to_lds_b128 %0, %1, off offset:16"
                   :: "v"(dst_off), "v"(src) : "memory");
#else
      *(h8*)(dst)     = *(const h8*)(src);
      *(h8*)(dst + 8) = *(const h8*)(src + 8);
#endif
      if (k0 + TK < K) __builtin_prefetch(src + TK, 0, 1);   // global_prefetch_b8
    }
    // ---- stage B tile (32 x 64) transposed into Bs[col][k] ----
    {
      const int r   = tid >> 3;                 // k row 0..31
      const int seg = (tid & 7) * 8;            // col segment
      const _Float16* src = B + (size_t)(k0 + r) * ldb + (bcol + seg);
      h8 bv = *(const h8*)(src);
#pragma unroll
      for (int i = 0; i < 8; ++i) Bs[(seg + i) * LDS_BS + r] = bv[i];
      if (k0 + TK < K) __builtin_prefetch(src + (size_t)TK * ldb, 0, 1);
    }
#ifdef __gfx1250__
    asm volatile("s_wait_asynccnt 0x0" ::: "memory");  // own async copies done
#endif
    __syncthreads();                                   // all waves' copies done

    // ---- fragment loads (contiguous ds_load_b128 pairs) ----
    Frag af[2], bf[2];
#pragma unroll
    for (int mi = 0; mi < 2; ++mi) {
      // A 16x32 f16 layout: lane<16 -> M=lane, K in {kbase..+7, 16+kbase..+7}
      const _Float16* rp = As + (wm * 32 + mi * 16 + laneIdx) * LDS_AS;
      af[mi].q[0] = *(const h8*)(rp + kbase);
      af[mi].q[1] = *(const h8*)(rp + 16 + kbase);
    }
#pragma unroll
    for (int ni = 0; ni < 2; ++ni) {
      // B 32x16 f16 layout: laneIdx -> N column, K in {kbase..+7, 16+kbase..+7}
      const _Float16* cp = Bs + (wn * 32 + ni * 16 + laneIdx) * LDS_BS + kbase;
      bf[ni].q[0] = *(const h8*)(cp);
      bf[ni].q[1] = *(const h8*)(cp + 16);
    }
#pragma unroll
    for (int mi = 0; mi < 2; ++mi)
#pragma unroll
      for (int ni = 0; ni < 2; ++ni)
        acc[mi][ni] = __builtin_amdgcn_wmma_f32_16x16x32_f16(
            false, af[mi].v, false, bf[ni].v, (short)0, acc[mi][ni], false, false);
    __syncthreads();
  }

  // ---- epilogue: C layout lanes 0-15 -> M=p, lanes 16-31 -> M=8+p ----
#pragma unroll
  for (int mi = 0; mi < 2; ++mi) {
#pragma unroll
    for (int ni = 0; ni < 2; ++ni) {
      const int r0 = brow + wm * 32 + mi * 16 + laneHalf * 8;
      const int c  = bcol + wn * 32 + ni * 16 + laneIdx;
#pragma unroll
      for (int p = 0; p < 8; ++p) {
        float v = acc[mi][ni][p] * alpha;
        if (bias) v += bias[r0 + p];
        C[(size_t)(r0 + p) * ldc + c] = v;
      }
    }
  }
}

// ========================= elementwise helpers =============================

__global__ void k_cvt_f16(const float* __restrict__ in, _Float16* __restrict__ out, int n) {
  int i = blockIdx.x * blockDim.x + threadIdx.x;
  if (i < n) out[i] = (_Float16)in[i];
}

__global__ void k_add_inplace(float* __restrict__ d, const float* __restrict__ delta, int n) {
  int i = blockIdx.x * blockDim.x + threadIdx.x;
  if (i < n) d[i] += delta[i];
}

__global__ void k_zero_f32(float* __restrict__ p, int n) {
  int i = blockIdx.x * blockDim.x + threadIdx.x;
  if (i < n) p[i] = 0.0f;
}

// posenc: cosE/sinE[n][j] = cos/sin( norm_kpt(n) . Wr[j] ),  j in [0,32)
__global__ void k_posenc(const float* __restrict__ kpts, const float* __restrict__ Wr,
                         float* __restrict__ cosE, float* __restrict__ sinE) {
  int idx = blockIdx.x * blockDim.x + threadIdx.x;
  if (idx >= cfg::N * 32) return;
  int j = idx & 31;
  int n = idx >> 5;
  float x = (kpts[n * 2 + 0] - 512.0f) / 716.8f;   // center=(512,384), scale=1024*0.7
  float y = (kpts[n * 2 + 1] - 384.0f) / 716.8f;
  float p = x * Wr[j * 2 + 0] + y * Wr[j * 2 + 1];
  cosE[idx] = cosf(p);
  sinE[idx] = sinf(p);
}

// (D,N) f32 -> head-major (H,N,HD) f16 with rotary applied.
// channel c = d*H + h;  rotate_half: even d -> -x[d+1], odd d -> x[d-1]
__global__ void k_rope_headT(const float* __restrict__ q,
                             const float* __restrict__ cosE, const float* __restrict__ sinE,
                             _Float16* __restrict__ out) {
  int idx = blockIdx.x * blockDim.x + threadIdx.x;
  if (idx >= cfg::D * cfg::N) return;
  int d = idx % cfg::HD;
  int n = (idx / cfg::HD) % cfg::N;
  int h = idx / (cfg::HD * cfg::N);
  float t = q[(size_t)(d * cfg::H + h) * cfg::N + n];
  float r = (d & 1) ? q[(size_t)((d - 1) * cfg::H + h) * cfg::N + n]
                    : -q[(size_t)((d + 1) * cfg::H + h) * cfg::N + n];
  float cs = cosE[n * 32 + (d >> 1)];
  float sn = sinE[n * 32 + (d >> 1)];
  out[idx] = (_Float16)(t * cs + r * sn);
}

// (D,N) f32 -> (D,N) f16 with rotary applied (for K operand).
__global__ void k_rope_dn(const float* __restrict__ k,
                          const float* __restrict__ cosE, const float* __restrict__ sinE,
                          _Float16* __restrict__ out) {
  int idx = blockIdx.x * blockDim.x + threadIdx.x;
  if (idx >= cfg::D * cfg::N) return;
  int c = idx / cfg::N;
  int n = idx % cfg::N;
  int d = c >> 2;       // H == 4
  int h = c & 3;
  float t = k[idx];
  float r = (d & 1) ? k[(size_t)((d - 1) * cfg::H + h) * cfg::N + n]
                    : -k[(size_t)((d + 1) * cfg::H + h) * cfg::N + n];
  float cs = cosE[n * 32 + (d >> 1)];
  float sn = sinE[n * 32 + (d >> 1)];
  out[idx] = (_Float16)(t * cs + r * sn);
}

// (D,N) f32 -> head-major (H,N,HD) f16 (no rotary; used for v always, q in cross layers)
__global__ void k_headT_cvt(const float* __restrict__ in, _Float16* __restrict__ out) {
  int idx = blockIdx.x * blockDim.x + threadIdx.x;
  if (idx >= cfg::D * cfg::N) return;
  int d = idx % cfg::HD;
  int n = (idx / cfg::HD) % cfg::N;
  int h = idx / (cfg::HD * cfg::N);
  out[idx] = (_Float16)in[(size_t)(d * cfg::H + h) * cfg::N + n];
}

// head-major (H,N,HD) f32 -> (D,N) f16  (merge heads back to channel layout)
__global__ void k_merge_heads(const float* __restrict__ in, _Float16* __restrict__ out) {
  int idx = blockIdx.x * blockDim.x + threadIdx.x;
  if (idx >= cfg::D * cfg::N) return;
  int c = idx / cfg::N;
  int n = idx % cfg::N;
  int d = c >> 2;
  int h = c & 3;
  out[idx] = (_Float16)in[((size_t)h * cfg::N + n) * cfg::HD + d];
}

// (D,N) f32 -> (N,D) f16  (plain transpose, for final mdesc0^T)
__global__ void k_transpose_cvt(const float* __restrict__ in, _Float16* __restrict__ out) {
  int idx = blockIdx.x * blockDim.x + threadIdx.x;
  if (idx >= cfg::D * cfg::N) return;
  int c = idx % cfg::D;
  int n = idx / cfg::D;
  out[idx] = (_Float16)in[(size_t)c * cfg::N + n];
}

// row softmax: scores (rows x cols f32) -> prob f16
__global__ __launch_bounds__(256) void k_softmax_row(const float* __restrict__ S,
                                                     _Float16* __restrict__ P, int cols) {
  __shared__ float red[256];
  const int row = blockIdx.x;
  const float* s = S + (size_t)row * cols;
  float m = -3.4e38f;
  for (int j = threadIdx.x; j < cols; j += 256) m = fmaxf(m, s[j]);
  red[threadIdx.x] = m; __syncthreads();
  for (int off = 128; off > 0; off >>= 1) {
    if (threadIdx.x < off) red[threadIdx.x] = fmaxf(red[threadIdx.x], red[threadIdx.x + off]);
    __syncthreads();
  }
  m = red[0]; __syncthreads();
  float sum = 0.0f;
  for (int j = threadIdx.x; j < cols; j += 256) sum += __expf(s[j] - m);
  red[threadIdx.x] = sum; __syncthreads();
  for (int off = 128; off > 0; off >>= 1) {
    if (threadIdx.x < off) red[threadIdx.x] += red[threadIdx.x + off];
    __syncthreads();
  }
  const float inv = 1.0f / red[0];
  _Float16* p = P + (size_t)row * cols;
  for (int j = threadIdx.x; j < cols; j += 256) p[j] = (_Float16)(__expf(s[j] - m) * inv);
}

// batchnorm stats per channel over N (B==1): stats[2c]=mean, stats[2c+1]=rstd
__global__ __launch_bounds__(256) void k_bn_stats(const float* __restrict__ hbuf,
                                                  float* __restrict__ stats) {
  __shared__ float s1[256], s2[256];
  const int c = blockIdx.x;
  const float* x = hbuf + (size_t)c * cfg::N;
  float a = 0.0f, b = 0.0f;
  for (int j = threadIdx.x; j < cfg::N; j += 256) { float v = x[j]; a += v; b += v * v; }
  s1[threadIdx.x] = a; s2[threadIdx.x] = b; __syncthreads();
  for (int off = 128; off > 0; off >>= 1) {
    if (threadIdx.x < off) { s1[threadIdx.x] += s1[threadIdx.x + off];
                             s2[threadIdx.x] += s2[threadIdx.x + off]; }
    __syncthreads();
  }
  if (threadIdx.x == 0) {
    float mean = s1[0] / cfg::N;
    float var  = s2[0] / cfg::N - mean * mean;
    stats[2 * c]     = mean;
    stats[2 * c + 1] = rsqrtf(var + 1e-5f);
  }
}

// bn apply + relu -> f16
__global__ void k_bn_apply(const float* __restrict__ hbuf, const float* __restrict__ stats,
                           const float* __restrict__ g, const float* __restrict__ b,
                           _Float16* __restrict__ out, int channels) {
  int idx = blockIdx.x * blockDim.x + threadIdx.x;
  if (idx >= channels * cfg::N) return;
  int c = idx / cfg::N;
  float v = g[c] * (hbuf[idx] - stats[2 * c]) * stats[2 * c + 1] + b[c];
  out[idx] = (_Float16)fmaxf(v, 0.0f);
}

// fill dustbin row/col of couplings with bin_score (read from device scalar)
__global__ void k_fill_bins(float* __restrict__ out, const float* __restrict__ alphap) {
  int i = blockIdx.x * blockDim.x + threadIdx.x;
  if (i >= cfg::SN) return;
  float a = *alphap;
  out[(size_t)2048 * cfg::SN + i] = a;
  out[(size_t)i * cfg::SN + 2048] = a;
}

// Sinkhorn: u[row] = log_mu(row) - lse_m(cpl[row,m] + v[m])
__global__ __launch_bounds__(256) void k_sink_u(const float* __restrict__ cpl,
                                                const float* __restrict__ v,
                                                float* __restrict__ u) {
  __shared__ float red[256];
  const int row = blockIdx.x;
  const float* c = cpl + (size_t)row * cfg::SN;
  float m = -3.4e38f;
  for (int j = threadIdx.x; j < cfg::SN; j += 256) m = fmaxf(m, c[j] + v[j]);
  red[threadIdx.x] = m; __syncthreads();
  for (int off = 128; off > 0; off >>= 1) {
    if (threadIdx.x < off) red[threadIdx.x] = fmaxf(red[threadIdx.x], red[threadIdx.x + off]);
    __syncthreads();
  }
  m = red[0]; __syncthreads();
  float sum = 0.0f;
  for (int j = threadIdx.x; j < cfg::SN; j += 256) sum += __expf(c[j] + v[j] - m);
  red[threadIdx.x] = sum; __syncthreads();
  for (int off = 128; off > 0; off >>= 1) {
    if (threadIdx.x < off) red[threadIdx.x] += red[threadIdx.x + off];
    __syncthreads();
  }
  if (threadIdx.x == 0) {
    float logmu = (row < 2048) ? cfg::NORM : (cfg::LOG_N + cfg::NORM);
    u[row] = logmu - (m + __logf(red[0]));
  }
}

// Sinkhorn: v[col] = log_nu(col) - lse_n(cpl[n,col] + u[n])
__global__ __launch_bounds__(256) void k_sink_v(const float* __restrict__ cpl,
                                                const float* __restrict__ u,
                                                float* __restrict__ v) {
  __shared__ float red[256];
  const int col = blockIdx.x;
  float m = -3.4e38f;
  for (int j = threadIdx.x; j < cfg::SN; j += 256)
    m = fmaxf(m, cpl[(size_t)j * cfg::SN + col] + u[j]);
  red[threadIdx.x] = m; __syncthreads();
  for (int off = 128; off > 0; off >>= 1) {
    if (threadIdx.x < off) red[threadIdx.x] = fmaxf(red[threadIdx.x], red[threadIdx.x + off]);
    __syncthreads();
  }
  m = red[0]; __syncthreads();
  float sum = 0.0f;
  for (int j = threadIdx.x; j < cfg::SN; j += 256)
    sum += __expf(cpl[(size_t)j * cfg::SN + col] + u[j] - m);
  red[threadIdx.x] = sum; __syncthreads();
  for (int off = 128; off > 0; off >>= 1) {
    if (threadIdx.x < off) red[threadIdx.x] += red[threadIdx.x + off];
    __syncthreads();
  }
  if (threadIdx.x == 0) {
    float lognu = (col < 2048) ? cfg::NORM : (cfg::LOG_N + cfg::NORM);
    v[col] = lognu - (m + __logf(red[0]));
  }
}

__global__ void k_final_add(float* __restrict__ out, const float* __restrict__ u,
                            const float* __restrict__ v) {
  int idx = blockIdx.x * blockDim.x + threadIdx.x;
  if (idx >= cfg::SN * cfg::SN) return;
  int r = idx / cfg::SN, c = idx % cfg::SN;
  out[idx] += u[r] + v[c] - cfg::NORM;
}

// =============================== host side =================================

static inline int cdiv(int a, int b) { return (a + b - 1) / b; }

static void gemm(hipStream_t s, const _Float16* A, int lda, const _Float16* B, int ldb,
                 float* C, int ldc, int M, int N, int K, float alpha, const float* bias) {
  dim3 g(N / TN, M / TM), b(256);
  gemm_wmma_f16<<<g, b, 0, s>>>(A, lda, B, ldb, C, ldc, M, N, K, alpha, bias);
}

extern "C" void kernel_launch(void* const* d_in, const int* in_sizes, int n_in,
                              void* d_out, int out_size, void* d_ws, size_t ws_size,
                              hipStream_t stream) {
  using namespace cfg;
  (void)in_sizes; (void)n_in; (void)out_size; (void)ws_size;

  const float* desc0   = (const float*)d_in[0];
  const float* desc1   = (const float*)d_in[1];
  const float* kpts0   = (const float*)d_in[2];
  const float* kpts1   = (const float*)d_in[3];
  const float* Wr      = (const float*)d_in[4];
  const float* proj_w  = (const float*)d_in[5];
  const float* proj_b  = (const float*)d_in[6];
  const float* merge_w = (const float*)d_in[7];
  const float* merge_b = (const float*)d_in[8];
  const float* mlp_w1  = (const float*)d_in[9];
  const float* mlp_b1  = (const float*)d_in[10];
  const float* bn_g    = (const float*)d_in[11];
  const float* bn_b    = (const float*)d_in[12];
  const float* mlp_w2  = (const float*)d_in[13];
  const float* mlp_b2  = (const float*)d_in[14];
  const float* fp_w    = (const float*)d_in[15];
  const float* fp_b    = (const float*)d_in[16];
  const float* binsc   = (const float*)d_in[17];
  float* out = (float*)d_out;

  // ---- workspace carve-up (aligned 256B) ----
  char* wp = (char*)d_ws;
  auto alloc = [&](size_t bytes) -> void* {
    void* r = wp; wp += (bytes + 255) & ~(size_t)255; return r;
  };
  const size_t DN = (size_t)D * N;
  _Float16* w16_proj  = (_Float16*)alloc(sizeof(_Float16) * (size_t)L * 3 * D * D);
  _Float16* w16_merge = (_Float16*)alloc(sizeof(_Float16) * (size_t)L * D * D);
  _Float16* w16_mlp1  = (_Float16*)alloc(sizeof(_Float16) * (size_t)L * 2 * D * 2 * D);
  _Float16* w16_mlp2  = (_Float16*)alloc(sizeof(_Float16) * (size_t)L * D * 2 * D);
  _Float16* w16_fp    = (_Float16*)alloc(sizeof(_Float16) * (size_t)D * D);
  float*    d0c   = (float*)alloc(sizeof(float) * DN);
  float*    d1c   = (float*)alloc(sizeof(float) * DN);
  _Float16* xa16  = (_Float16*)alloc(sizeof(_Float16) * DN);
  _Float16* xb16  = (_Float16*)alloc(sizeof(_Float16) * DN);
  float*    qf    = (float*)alloc(sizeof(float) * DN);
  float*    kf    = (float*)alloc(sizeof(float) * DN);
  float*    vf    = (float*)alloc(sizeof(float) * DN);
  _Float16* qT16  = (_Float16*)alloc(sizeof(_Float16) * DN);   // (H,N,HD)
  _Float16* k16   = (_Float16*)alloc(sizeof(_Float16) * DN);   // (D,N)
  _Float16* vT16  = (_Float16*)alloc(sizeof(_Float16) * DN);   // (H,N,HD)
  float*    sc    = (float*)alloc(sizeof(float) * (size_t)N * N);       // per-head scores
  _Float16* prob16= (_Float16*)alloc(sizeof(_Float16) * (size_t)N * N);
  float*    outT  = (float*)alloc(sizeof(float) * DN);         // (H,N,HD)
  _Float16* attn16= (_Float16*)alloc(sizeof(_Float16) * DN);   // (D,N)
  float*    msg   = (float*)alloc(sizeof(float) * DN);
  _Float16* y16   = (_Float16*)alloc(sizeof(_Float16) * 2 * DN);   // concat [x; msg]
  float*    hbuf  = (float*)alloc(sizeof(float) * 2 * DN);
  _Float16* hn16  = (_Float16*)alloc(sizeof(_Float16) * 2 * DN);
  float*    delta0= (float*)alloc(sizeof(float) * DN);
  float*    delta1= (float*)alloc(sizeof(float) * DN);
  float*    mdesc0= (float*)alloc(sizeof(float) * DN);
  float*    mdesc1= (float*)alloc(sizeof(float) * DN);
  _Float16* md0T16= (_Float16*)alloc(sizeof(_Float16) * DN);   // (N,D)
  float*    cos0  = (float*)alloc(sizeof(float) * (size_t)N * 32);
  float*    sin0  = (float*)alloc(sizeof(float) * (size_t)N * 32);
  float*    cos1  = (float*)alloc(sizeof(float) * (size_t)N * 32);
  float*    sin1  = (float*)alloc(sizeof(float) * (size_t)N * 32);
  float*    stats = (float*)alloc(sizeof(float) * 2 * 2 * D);
  float*    uv    = (float*)alloc(sizeof(float) * 2 * SN);
  float* uvec = uv, * vvec = uv + SN;

  const int EB = 256;
  auto grid1 = [&](int n) { return dim3(cdiv(n, EB)); };

  // ---- one-time per call: f16 weights, working copies, pos enc ----
  k_cvt_f16<<<grid1(L*3*D*D), EB, 0, stream>>>(proj_w,  w16_proj,  L*3*D*D);
  k_cvt_f16<<<grid1(L*D*D),   EB, 0, stream>>>(merge_w, w16_merge, L*D*D);
  k_cvt_f16<<<grid1(L*4*D*D), EB, 0, stream>>>(mlp_w1,  w16_mlp1,  L*4*D*D);
  k_cvt_f16<<<grid1(L*2*D*D), EB, 0, stream>>>(mlp_w2,  w16_mlp2,  L*2*D*D);
  k_cvt_f16<<<grid1(D*D),     EB, 0, stream>>>(fp_w,    w16_fp,    D*D);
  (void)hipMemcpyAsync(d0c, desc0, sizeof(float) * DN, hipMemcpyDeviceToDevice, stream);
  (void)hipMemcpyAsync(d1c, desc1, sizeof(float) * DN, hipMemcpyDeviceToDevice, stream);
  k_posenc<<<grid1(N*32), EB, 0, stream>>>(kpts0, Wr, cos0, sin0);
  k_posenc<<<grid1(N*32), EB, 0, stream>>>(kpts1, Wr, cos1, sin1);

  // ---- 10 attentional GNN layers ----
  for (int i = 0; i < L; ++i) {
    const bool self = (i % 2 == 0);
    k_cvt_f16<<<grid1((int)DN), EB, 0, stream>>>(d0c, xa16, (int)DN);
    k_cvt_f16<<<grid1((int)DN), EB, 0, stream>>>(d1c, xb16, (int)DN);

    const _Float16* pw = w16_proj + (size_t)i * 3 * D * D;
    const float*    pb = proj_b   + (size_t)i * 3 * D;

    for (int t = 0; t < 2; ++t) {
      const _Float16* x16   = t ? xb16 : xa16;
      const _Float16* src16 = self ? x16 : (t ? xa16 : xb16);
      float* dcur  = t ? d1c : d0c;
      float* delta = t ? delta1 : delta0;
      const float* cE = t ? cos1 : cos0;
      const float* sE = t ? sin1 : sin0;

      // q/k/v projections (WMMA GEMMs, K=256)
      gemm(stream, pw + 0 * D * D, D, x16,   N, qf, N, D, N, D, 1.0f, pb + 0 * D);
      gemm(stream, pw + 1 * D * D, D, src16, N, kf, N, D, N, D, 1.0f, pb + 1 * D);
      gemm(stream, pw + 2 * D * D, D, src16, N, vf, N, D, N, D, 1.0f, pb + 2 * D);

      if (self) {
        k_rope_headT<<<grid1((int)DN), EB, 0, stream>>>(qf, cE, sE, qT16);
        k_rope_dn  <<<grid1((int)DN), EB, 0, stream>>>(kf, cE, sE, k16);
      } else {
        k_headT_cvt<<<grid1((int)DN), EB, 0, stream>>>(qf, qT16);
        k_cvt_f16  <<<grid1((int)DN), EB, 0, stream>>>(kf, k16, (int)DN);
      }
      k_headT_cvt<<<grid1((int)DN), EB, 0, stream>>>(vf, vT16);

      // per-head attention
      for (int h = 0; h < H; ++h) {
        // scores = q^T k / 8  (N x N, K=64)
        gemm(stream, qT16 + (size_t)h * N * HD, HD, k16 + (size_t)h * N, H * N,
             sc, N, N, N, HD, 0.125f, nullptr);
        k_softmax_row<<<N, 256, 0, stream>>>(sc, prob16, N);
        // out = prob @ v  (N x 64, K=2048)
        gemm(stream, prob16, N, vT16 + (size_t)h * N * HD, HD,
             outT + (size_t)h * N * HD, HD, N, HD, N, 1.0f, nullptr);
      }
      k_merge_heads<<<grid1((int)DN), EB, 0, stream>>>(outT, attn16);

      // message = merge_w @ attn + mb
      gemm(stream, w16_merge + (size_t)i * D * D, D, attn16, N, msg, N,
           D, N, D, 1.0f, merge_b + (size_t)i * D);

      // y = concat([x, message]) -> f16
      k_cvt_f16<<<grid1((int)DN), EB, 0, stream>>>(dcur, y16,      (int)DN);
      k_cvt_f16<<<grid1((int)DN), EB, 0, stream>>>(msg,  y16 + DN, (int)DN);

      // MLP: h = w1 @ y + b1 ; batchnorm + relu ; delta = w2 @ h + b2
      gemm(stream, w16_mlp1 + (size_t)i * 4 * D * D, 2 * D, y16, N, hbuf, N,
           2 * D, N, 2 * D, 1.0f, mlp_b1 + (size_t)i * 2 * D);
      k_bn_stats<<<2 * D, 256, 0, stream>>>(hbuf, stats);
      k_bn_apply<<<grid1(2 * (int)DN), EB, 0, stream>>>(
          hbuf, stats, bn_g + (size_t)i * 2 * D, bn_b + (size_t)i * 2 * D, hn16, 2 * D);
      gemm(stream, w16_mlp2 + (size_t)i * 2 * D * D, 2 * D, hn16, N, delta, N,
           D, N, 2 * D, 1.0f, mlp_b2 + (size_t)i * D);
    }
    // residual after both deltas (cross layers read both originals)
    k_add_inplace<<<grid1((int)DN), EB, 0, stream>>>(d0c, delta0, (int)DN);
    k_add_inplace<<<grid1((int)DN), EB, 0, stream>>>(d1c, delta1, (int)DN);
  }

  // ---- final projection + scores into d_out (ldc = 2049) ----
  k_cvt_f16<<<grid1((int)DN), EB, 0, stream>>>(d0c, xa16, (int)DN);
  k_cvt_f16<<<grid1((int)DN), EB, 0, stream>>>(d1c, xb16, (int)DN);
  gemm(stream, w16_fp, D, xa16, N, mdesc0, N, D, N, D, 1.0f, fp_b);
  gemm(stream, w16_fp, D, xb16, N, mdesc1, N, D, N, D, 1.0f, fp_b);
  k_transpose_cvt<<<grid1((int)DN), EB, 0, stream>>>(mdesc0, md0T16);
  k_cvt_f16<<<grid1((int)DN), EB, 0, stream>>>(mdesc1, k16, (int)DN);
  gemm(stream, md0T16, D, k16, N, out, SN, N, N, D, 0.0625f, nullptr);  // / sqrt(256)
  k_fill_bins<<<grid1(SN), EB, 0, stream>>>(out, binsc);

  // ---- Sinkhorn (20 iterations) ----
  k_zero_f32<<<grid1(2 * SN), EB, 0, stream>>>(uv, 2 * SN);
  for (int it = 0; it < 20; ++it) {
    k_sink_u<<<SN, 256, 0, stream>>>(out, vvec, uvec);
    k_sink_v<<<SN, 256, 0, stream>>>(out, uvec, vvec);
  }
  k_final_add<<<grid1(SN * SN), EB, 0, stream>>>(out, uvec, vvec);
}
